// Head_34351148433932
// MI455X (gfx1250) — matrix operand order repaired and checked by reading, not compiled, not run on hardware
//
#include <hip/hip_runtime.h>
#include <stdint.h>

typedef __attribute__((ext_vector_type(16))) __bf16 v16bf;
typedef __attribute__((ext_vector_type(2)))  __bf16 v2bf;
typedef __attribute__((ext_vector_type(8)))  float  v8f;

#define WMMA_BF16(A, B, C) __builtin_amdgcn_wmma_f32_16x16x32_bf16(false, (A), false, (B), (short)0, (C), false, false)

// workspace layout (bytes)
#define WP_OFF  0                       // packed weights: 3*32*4*32*8 dwords = 384 KB
#define Q_OFF   (524288)                // Q bf16 [65536][64] = 8 MB
#define K_OFF   (Q_OFF + 8388608)       // K bf16 [65536][64]
#define V_OFF   (K_OFF + 8388608)       // V^T bf16 [256][64][256]

// ---- float -> bf16 conversion (RNE); hw packed cvt if the toolchain has it ----
#if __has_builtin(__builtin_amdgcn_cvt_pk_bf16_f32)
__device__ __forceinline__ v2bf cvt2(float lo, float hi) {
    return __builtin_amdgcn_cvt_pk_bf16_f32(lo, hi);
}
#else
__device__ __forceinline__ __bf16 f2bf_manual(float f) {
    uint32_t u = __float_as_uint(f);
    u += 0x7FFFu + ((u >> 16) & 1u);
    unsigned short b = (unsigned short)(u >> 16);
    return __builtin_bit_cast(__bf16, b);
}
__device__ __forceinline__ v2bf cvt2(float lo, float hi) {
    v2bf r; r[0] = f2bf_manual(lo); r[1] = f2bf_manual(hi); return r;
}
#endif
__device__ __forceinline__ __bf16 f2bf(float f) { v2bf t = cvt2(f, f); return t[0]; }

union BV16 { uint4 u[2]; v16bf v; };
union A16  { v2bf  p[8]; v16bf v; };

__device__ __forceinline__ v16bf load_bf16x16(const __bf16* p) {
    BV16 t;
    t.u[0] = *(const uint4*)(p);
    t.u[1] = *(const uint4*)(p + 8);
    return t.v;
}

// ---------------------------------------------------------------------------
// Kernel 0: repack Wq/Wk/Wv [1024][64] fp32 -> bf16 in WMMA B-matrix layout.
// Chunk (m,kt) is 4 KB contiguous: [nt][lane][8 dwords];
// dword d = pack(W[k][n], W[k+1][n]), k = kb + 2*(d&3) + (d>=4 ? 16 : 0),
// kb = kt*32 + (lane>=16 ? 8 : 0), n = nt*16 + (lane&15).
// ---------------------------------------------------------------------------
__global__ void repack_w(const float* __restrict__ Wq, const float* __restrict__ Wk,
                         const float* __restrict__ Wv, __bf16* __restrict__ Wp) {
    int idx = blockIdx.x * blockDim.x + threadIdx.x;   // dword index < 98304
    int d    = idx & 7;
    int lane = (idx >> 3) & 31;
    int nt   = (idx >> 8) & 3;
    int kt   = (idx >> 10) & 31;
    int m    = idx >> 15;                              // 0..2
    const float* W = (m == 0) ? Wq : (m == 1) ? Wk : Wv;
    int k = kt * 32 + ((lane & 16) ? 8 : 0) + (d & 3) * 2 + ((d & 4) ? 16 : 0);
    int n = nt * 16 + (lane & 15);
    v2bf pk = cvt2(W[k * 64 + n], W[(k + 1) * 64 + n]);
    Wp[idx * 2]     = pk[0];
    Wp[idx * 2 + 1] = pk[1];
}

// ---------------------------------------------------------------------------
// Kernel 1: fused QKV projection, 256 threads = 8 waves x 16 rows.
// x streamed from HBM once (software-pipelined float4 loads); weights staged
// per K-step into a double-buffered LDS tile (12 KB) shared by all 8 waves;
// B-tiles then come from ds_load_b128 instead of per-wave L2 round trips.
// Q,K stored row-major bf16 [row][64]; V stored transposed bf16 [b][h][t].
// ---------------------------------------------------------------------------
__global__ void __launch_bounds__(256) qkv_proj(
    const float* __restrict__ x, const __bf16* __restrict__ Wp,
    __bf16* __restrict__ Qb, __bf16* __restrict__ Kb, __bf16* __restrict__ VT)
{
    __shared__ __bf16 wbuf[2][3 * 2048];               // 2 x 12 KB

    const int t       = threadIdx.x;                   // 0..255
    const int lane    = t & 31;
    const int wave    = t >> 5;                        // 0..7
    const int tile    = blockIdx.x * 8 + wave;         // 0..4095
    const int rowbase = tile * 16;
    const int lane_lo = lane & 15;
    const int g       = lane >> 4;

    v8f zero = {};
    v8f acc[3][4];
#pragma unroll
    for (int m = 0; m < 3; ++m)
#pragma unroll
        for (int nt = 0; nt < 4; ++nt) acc[m][nt] = zero;

    // A-matrix addressing: row = lane&15; K chunks {koff..+7, 16+koff..+7}
    const int arow = rowbase + lane_lo;
    const int koff = g * 8;
    const float* xr = x + (size_t)arow * 1024 + koff;

    const uint4* wsrc = (const uint4*)Wp;              // 256 uint4 per (m,kt) chunk

    // ---- prologue: stage kt=0 weights, start kt=0 x loads
    {
        uint4 g0 = wsrc[(size_t)(0 * 32 + 0) * 256 + t];
        uint4 g1 = wsrc[(size_t)(1 * 32 + 0) * 256 + t];
        uint4 g2 = wsrc[(size_t)(2 * 32 + 0) * 256 + t];
        uint4* dst = (uint4*)wbuf[0];
        dst[0 * 256 + t] = g0;
        dst[1 * 256 + t] = g1;
        dst[2 * 256 + t] = g2;
    }
    float4 f0 = *(const float4*)(xr);
    float4 f1 = *(const float4*)(xr + 4);
    float4 f2 = *(const float4*)(xr + 16);
    float4 f3 = *(const float4*)(xr + 20);
    __syncthreads();

    for (int kt = 0; kt < 32; ++kt) {
        const __bf16* cur = wbuf[kt & 1];
        __bf16* nxt = wbuf[(kt + 1) & 1];

        // issue next K-step's weight loads now; consumed at bottom of iteration
        uint4 g0, g1, g2;
        if (kt < 31) {
            g0 = wsrc[(size_t)(0 * 32 + kt + 1) * 256 + t];
            g1 = wsrc[(size_t)(1 * 32 + kt + 1) * 256 + t];
            g2 = wsrc[(size_t)(2 * 32 + kt + 1) * 256 + t];
        }

        // convert current A tile (bf16 pairs)
        A16 au;
        au.p[0] = cvt2(f0.x, f0.y); au.p[1] = cvt2(f0.z, f0.w);
        au.p[2] = cvt2(f1.x, f1.y); au.p[3] = cvt2(f1.z, f1.w);
        au.p[4] = cvt2(f2.x, f2.y); au.p[5] = cvt2(f2.z, f2.w);
        au.p[6] = cvt2(f3.x, f3.y); au.p[7] = cvt2(f3.z, f3.w);
        v16bf a = au.v;

        // issue next iteration's x loads
        if (kt < 31) {
            const float* xp = xr + (kt + 1) * 32;
            f0 = *(const float4*)(xp);
            f1 = *(const float4*)(xp + 4);
            f2 = *(const float4*)(xp + 16);
            f3 = *(const float4*)(xp + 20);
        }

        // compute from LDS-staged weights
#pragma unroll
        for (int m = 0; m < 3; ++m) {
            const __bf16* bp = cur + m * 2048 + (size_t)lane * 16;
            v16bf b0 = load_bf16x16(bp);
            v16bf b1 = load_bf16x16(bp + 512);
            v16bf b2 = load_bf16x16(bp + 1024);
            v16bf b3 = load_bf16x16(bp + 1536);
            acc[m][0] = WMMA_BF16(a, b0, acc[m][0]);
            acc[m][1] = WMMA_BF16(a, b1, acc[m][1]);
            acc[m][2] = WMMA_BF16(a, b2, acc[m][2]);
            acc[m][3] = WMMA_BF16(a, b3, acc[m][3]);
        }

        // write next K-step's weights to the alternate buffer
        if (kt < 31) {
            uint4* dst = (uint4*)nxt;
            dst[0 * 256 + t] = g0;
            dst[1 * 256 + t] = g1;
            dst[2 * 256 + t] = g2;
        }
        __syncthreads();
    }

    // C/D layout: VGPR r -> row (r + 8*g), col = nt*16 + lane_lo
#pragma unroll
    for (int nt = 0; nt < 4; ++nt) {
        const int col = nt * 16 + lane_lo;
#pragma unroll
        for (int r = 0; r < 8; ++r) {
            const int row = rowbase + r + 8 * g;
            Qb[(size_t)row * 64 + col] = f2bf(acc[0][nt][r]);
            Kb[(size_t)row * 64 + col] = f2bf(acc[1][nt][r]);
            const int bb = row >> 8, tt = row & 255;
            VT[((size_t)(bb * 64 + col)) * 256 + tt] = f2bf(acc[2][nt][r]);
        }
    }
}

// ---------------------------------------------------------------------------
// Kernel 2: flash attention. One wave owns a 16-query tile of one batch,
// streams 32 keys/iteration: 4 WMMAs for S = Q K^T, online softmax,
// P -> LDS -> A-layout, 4 WMMAs for O += P V. All global loads issued at the
// top of the iteration so softmax VALU work hides their latency.
// ---------------------------------------------------------------------------
__global__ void __launch_bounds__(128) attn(
    const __bf16* __restrict__ Qb, const __bf16* __restrict__ Kb,
    const __bf16* __restrict__ VT, float* __restrict__ out)
{
    __shared__ __bf16 plds[4][16 * 32];
    const int lane    = threadIdx.x & 31;
    const int wave    = threadIdx.x >> 5;
    const int tile    = blockIdx.x * 4 + wave;     // 0..4095
    const int b       = tile >> 4;
    const int qt      = tile & 15;
    const int lane_lo = lane & 15;
    const int g       = lane >> 4;
    __bf16* lds = plds[wave];

    // Q tile in A-layout: A0 covers h 0..31, A1 covers h 32..63
    const __bf16* qp = Qb + (size_t)(b * 256 + qt * 16 + lane_lo) * 64 + g * 16;
    v16bf A0, A1;
    {
        BV16 t0, t1;
        t0.u[0] = *(const uint4*)(qp);
        t0.u[1] = *(const uint4*)(qp + 8);
        t1.u[0] = *(const uint4*)(qp + 32);
        t1.u[1] = *(const uint4*)(qp + 40);
        A0 = t0.v; A1 = t1.v;
    }

    v8f o0 = {}, o1 = {}, o2 = {}, o3 = {};
    float m[8], l[8];
#pragma unroll
    for (int r = 0; r < 8; ++r) { m[r] = -1e30f; l[r] = 0.0f; }

    const int   jmax  = (qt * 16 + 15) >> 5;
    const float SCALE = 0.03125f;                  // C^-0.5, C = 1024
    const float L2E   = 1.4426950408889634f;

    for (int j = 0; j <= jmax; ++j) {
        // ---- issue all global loads for this iteration up front
        const int key = j * 32 + lane_lo;
        const __bf16* kp0 = Kb + (size_t)(b * 256 + key) * 64 + g * 16;
        const __bf16* kp1 = Kb + (size_t)(b * 256 + key + 16) * 64 + g * 16;
        v16bf K00 = load_bf16x16(kp0);
        v16bf K01 = load_bf16x16(kp0 + 32);
        v16bf K10 = load_bf16x16(kp1);
        v16bf K11 = load_bf16x16(kp1 + 32);
        const __bf16* vp = VT + (size_t)(b * 64 + lane_lo) * 256 + j * 32 + g * 16;
        v16bf V0 = load_bf16x16(vp);
        v16bf V1 = load_bf16x16(vp + 16 * 256);
        v16bf V2 = load_bf16x16(vp + 32 * 256);
        v16bf V3 = load_bf16x16(vp + 48 * 256);

        // ---- S = Q K^T (two 16x16 score tiles: keys j*32..+15 and +16..+31)
        v8f s0 = {}, s1 = {};
        s0 = WMMA_BF16(A0, K00, s0);
        s0 = WMMA_BF16(A1, K01, s0);
        s1 = WMMA_BF16(A0, K10, s1);
        s1 = WMMA_BF16(A1, K11, s1);

        // ---- online softmax (row = r + 8*g, col = lane_lo within 16-lane half)
        float corr[8];
#pragma unroll
        for (int r = 0; r < 8; ++r) {
            const int qrow = qt * 16 + r + 8 * g;
            const int k0   = j * 32 + lane_lo;
            float a0 = (k0      <= qrow) ? s0[r] * SCALE : -1e30f;
            float a1 = (k0 + 16 <= qrow) ? s1[r] * SCALE : -1e30f;
            float mx = fmaxf(a0, a1);
            mx = fmaxf(mx, __shfl_xor(mx, 1));
            mx = fmaxf(mx, __shfl_xor(mx, 2));
            mx = fmaxf(mx, __shfl_xor(mx, 4));
            mx = fmaxf(mx, __shfl_xor(mx, 8));
            const float mn = fmaxf(m[r], mx);
            const float p0 = exp2f((a0 - mn) * L2E);
            const float p1 = exp2f((a1 - mn) * L2E);
            float rs = p0 + p1;
            rs += __shfl_xor(rs, 1);
            rs += __shfl_xor(rs, 2);
            rs += __shfl_xor(rs, 4);
            rs += __shfl_xor(rs, 8);
            const float cr = exp2f((m[r] - mn) * L2E);
            l[r] = l[r] * cr + rs;
            m[r] = mn;
            corr[r] = cr;
            // write P (bf16) to LDS, row-major [16][32]
            lds[(r + 8 * g) * 32 + lane_lo]      = f2bf(p0);
            lds[(r + 8 * g) * 32 + 16 + lane_lo] = f2bf(p1);
        }
#pragma unroll
        for (int r = 0; r < 8; ++r) {
            o0[r] *= corr[r]; o1[r] *= corr[r]; o2[r] *= corr[r]; o3[r] *= corr[r];
        }

        // same-wave LDS RAW: DS ops are in-order per wave; wait for stores
        asm volatile("s_wait_dscnt 0" ::: "memory");

        // ---- read P back in A-layout [16 q, 32 s]
        v16bf P;
        {
            const __bf16* pp = lds + lane_lo * 32 + g * 16;
            BV16 t;
            t.u[0] = *(const uint4*)(pp);
            t.u[1] = *(const uint4*)(pp + 8);
            P = t.v;
        }

        // ---- O += P V : B-tiles contiguous along s thanks to V^T storage
        o0 = WMMA_BF16(P, V0, o0);
        o1 = WMMA_BF16(P, V1, o1);
        o2 = WMMA_BF16(P, V2, o2);
        o3 = WMMA_BF16(P, V3, o3);
    }

    // ---- epilogue: O / l, store fp32 [b][t][h]
    float inv[8];
#pragma unroll
    for (int r = 0; r < 8; ++r) inv[r] = 1.0f / l[r];
    float* op = out + (size_t)(b * 256 + qt * 16) * 64;
#pragma unroll
    for (int r = 0; r < 8; ++r) {
        const int row = r + 8 * g;
        op[(size_t)row * 64 +  0 + lane_lo] = o0[r] * inv[r];
        op[(size_t)row * 64 + 16 + lane_lo] = o1[r] * inv[r];
        op[(size_t)row * 64 + 32 + lane_lo] = o2[r] * inv[r];
        op[(size_t)row * 64 + 48 + lane_lo] = o3[r] * inv[r];
    }
}

extern "C" void kernel_launch(void* const* d_in, const int* in_sizes, int n_in,
                              void* d_out, int out_size, void* d_ws, size_t ws_size,
                              hipStream_t stream) {
    const float* x  = (const float*)d_in[0];
    const float* Wq = (const float*)d_in[1];
    const float* Wk = (const float*)d_in[2];
    const float* Wv = (const float*)d_in[3];

    __bf16* Wp = (__bf16*)((char*)d_ws + WP_OFF);
    __bf16* Qb = (__bf16*)((char*)d_ws + Q_OFF);
    __bf16* Kb = (__bf16*)((char*)d_ws + K_OFF);
    __bf16* VT = (__bf16*)((char*)d_ws + V_OFF);

    repack_w<<<384, 256, 0, stream>>>(Wq, Wk, Wv, Wp);          // 98304 dwords
    qkv_proj<<<512, 256, 0, stream>>>(x, Wp, Qb, Kb, VT);       // 512 blocks x 128 rows
    attn<<<1024, 128, 0, stream>>>(Qb, Kb, VT, (float*)d_out);  // 4096 q-tiles
}
